// GCN_6219112645195
// MI455X (gfx1250) — matrix operand order repaired and checked by reading, not compiled, not run on hardware
//
#include <hip/hip_runtime.h>

// ---------------------------------------------------------------------------
// GCN forward for MI455X (gfx1250).
// Dense GEMMs: v_wmma_f32_16x16x32_bf16, activations pre-converted to bf16,
// weights pre-packed into per-lane fragment order => inner loop is
// 4x global_load_b128 + v_wmma only. Scatter phases: hw f32/u32 atomics.
// ---------------------------------------------------------------------------

typedef __attribute__((ext_vector_type(16))) __bf16 v16bf;
typedef __attribute__((ext_vector_type(8)))  __bf16 v8bf;
typedef __attribute__((ext_vector_type(8)))  float  v8f;

union Vfrag { v16bf v; v8bf h[2]; };

#define HID 64
#define GG  512
#define EPSI 1e-5f
#define SELU_SCALE 1.0507009873554805f
#define SELU_ALPHA 1.6732632423543772f

__device__ __forceinline__ __bf16 f2bf(float f) {
    union { float f; unsigned u; } in; in.f = f;
    unsigned u = in.u;
    unsigned r = (u + 0x7FFFu + ((u >> 16) & 1u)) >> 16;   // round-nearest-even
    union { unsigned short s; __bf16 b; } out; out.s = (unsigned short)r;
    return out.b;
}

__device__ __forceinline__ float bf2f(__bf16 b) {
    union { __bf16 b; unsigned short s; } in; in.b = b;
    union { unsigned u; float f; } out; out.u = ((unsigned)in.s) << 16;
    return out.f;
}

// ---------------------------------------------------------------------------
// WMMA GEMM: C[M x Nd] = A[M x K] @ B[K x Nd] (+bias)(+relu).
// A: row-major bf16. Bp: bf16 pre-packed in fragment order (see pack_Bfrag).
// One wave computes a 16-row x 64-col strip (4 accumulators, A frag reused
// 4x per k-step). Block = 4 waves = 64 rows. grid = (ceil(M/64), Nd/64).
// flags: bit0 = add bias, bit1 = relu.
// ---------------------------------------------------------------------------
__global__ void wmma_gemm(const __bf16* __restrict__ A, const __bf16* __restrict__ Bp,
                          const float* __restrict__ bias, float* __restrict__ C,
                          int M, int Nd, int K, int flags) {
    const int wave = threadIdx.x >> 5;
    const int lane = threadIdx.x & 31;
    const int l16  = lane & 15;
    const int hi   = lane >> 4;
    const int tm   = blockIdx.x * 4 + wave;      // 16-row tile index
    if (tm * 16 >= M) return;                    // wave-uniform: EXEC stays all-1
    const int cg   = blockIdx.y;                 // 64-column group
    const int kb_n = K >> 5;
    const int m    = tm * 16 + l16;

    v8f acc[4] = {v8f{}, v8f{}, v8f{}, v8f{}};
    const __bf16* Arow = A + (size_t)m * K + 16 * hi;
    const __bf16* Bcol = Bp + ((size_t)(cg * 4) * kb_n * 32 + lane) * 16;

    for (int kb = 0; kb < kb_n; ++kb) {
        Vfrag a;
        a.h[0] = *(const v8bf*)(Arow);           // k = kb*32 + 16*hi + [0..7]
        a.h[1] = *(const v8bf*)(Arow + 8);       // k = kb*32 + 16*hi + 8 + [0..7]
        Arow += 32;
        #pragma unroll
        for (int t = 0; t < 4; ++t) {
            const __bf16* bp = Bcol + ((size_t)t * kb_n + kb) * 32 * 16;
            Vfrag b;
            b.h[0] = *(const v8bf*)(bp);
            b.h[1] = *(const v8bf*)(bp + 8);
            acc[t] = __builtin_amdgcn_wmma_f32_16x16x32_bf16(
                         false, a.v, false, b.v, (short)0, acc[t], false, false);
        }
    }

    #pragma unroll
    for (int t = 0; t < 4; ++t) {
        const int n  = (cg * 4 + t) * 16 + l16;
        const float bv = (flags & 1) ? bias[n] : 0.0f;
        #pragma unroll
        for (int r = 0; r < 8; ++r) {            // C: VGPR r -> row (r + 8*hi)
            const int mr = tm * 16 + r + 8 * hi;
            float v = acc[t][r] + bv;
            if (flags & 2) v = fmaxf(v, 0.0f);
            C[(size_t)mr * Nd + n] = v;
        }
    }
}

// Pack W[K x Nd] (row-major f32) into WMMA B-fragment order bf16:
// element (tn, kb, lane, j) <- W[(kb*32 + 16*(lane>>4) + j) * Nd + tn*16 + (lane&15)]
__global__ void pack_Bfrag(const float* __restrict__ W, __bf16* __restrict__ Bp,
                           int K, int Nd) {
    int t = blockIdx.x * blockDim.x + threadIdx.x;
    if (t >= K * Nd) return;
    const int kb_n = K >> 5;
    int j    = t & 15;
    int lane = (t >> 4) & 31;
    int idx  = t >> 9;
    int kb   = idx % kb_n;
    int tn   = idx / kb_n;
    int k = kb * 32 + 16 * (lane >> 4) + j;
    int n = tn * 16 + (lane & 15);
    Bp[t] = f2bf(W[(size_t)k * Nd + n]);
}

__global__ void cvt_f32_bf16(const float* __restrict__ src, __bf16* __restrict__ dst, int n) {
    int t = blockIdx.x * blockDim.x + threadIdx.x;
    if (t < n) dst[t] = f2bf(src[t]);
}

// --------------------------- utility kernels -------------------------------
__global__ void fill_f32(float* __restrict__ p, float v, int n) {
    int t = blockIdx.x * blockDim.x + threadIdx.x;
    if (t < n) p[t] = v;
}

__global__ void deg_add(const int* __restrict__ col, float* __restrict__ deg, int E) {
    int t = blockIdx.x * blockDim.x + threadIdx.x;
    if (t < E) atomicAdd(&deg[col[t]], 1.0f);
}

__global__ void deg_rsqrt(float* __restrict__ d, int n) {
    int t = blockIdx.x * blockDim.x + threadIdx.x;
    if (t < n) d[t] = rsqrtf(d[t]);
}

// ------------- scatter: sum of normalized neighbor features ----------------
__global__ void scatter_add(const float* __restrict__ src, const int* __restrict__ row,
                            const int* __restrict__ col, const float* __restrict__ dinv,
                            float* __restrict__ dst, int E, int Nn) {
    int t = blockIdx.x * blockDim.x + threadIdx.x;
    int total = (E + Nn) * HID;
    if (t >= total) return;
    int e = t >> 6, f = t & 63;
    int r, c;
    if (e < E) { r = row[e]; c = col[e]; } else { r = c = e - E; }
    float nrm = dinv[r] * dinv[c];
    atomicAdd(&dst[(size_t)c * HID + f], src[(size_t)r * HID + f] * nrm);
}

// ------------- SELU(x + bias) in place -------------------------------------
__global__ void selu_bias(float* __restrict__ z, const float* __restrict__ b, int total) {
    int t = blockIdx.x * blockDim.x + threadIdx.x;
    if (t >= total) return;
    float v = z[t] + b[t & (HID - 1)];
    z[t] = SELU_SCALE * (v > 0.0f ? v : SELU_ALPHA * (__expf(v) - 1.0f));
}

// ------------- scatter max via monotone uint key (one hw atomic) -----------
__global__ void scatter_max(const float* __restrict__ src, const int* __restrict__ row,
                            const int* __restrict__ col, unsigned* __restrict__ keys,
                            int E, int Nn) {
    int t = blockIdx.x * blockDim.x + threadIdx.x;
    int total = (E + Nn) * HID;
    if (t >= total) return;
    int e = t >> 6, f = t & 63;
    int r, c;
    if (e < E) { r = row[e]; c = col[e]; } else { r = c = e - E; }
    unsigned u = __float_as_uint(src[(size_t)r * HID + f]);
    unsigned k = ((int)u >= 0) ? (u | 0x80000000u) : ~u;   // order-preserving map
    atomicMax(&keys[(size_t)c * HID + f], k);
}

// decode max-keys directly into bf16 (feeds both downstream WMMA GEMMs)
__global__ void decode_max_bf16(const unsigned* __restrict__ keys,
                                __bf16* __restrict__ h, int total) {
    int t = blockIdx.x * blockDim.x + threadIdx.x;
    if (t >= total) return;
    unsigned k = keys[t];
    float f = (k & 0x80000000u) ? __uint_as_float(k ^ 0x80000000u)
                                : __uint_as_float(~k);
    h[t] = f2bf(f);
}

// ------------- row softmax (64 wide) accumulated into R, wave per row ------
__global__ void softmax_acc(const float* __restrict__ z, float* __restrict__ R, int Nn) {
    int wid  = (blockIdx.x * blockDim.x + threadIdx.x) >> 5;
    int lane = threadIdx.x & 31;
    if (wid >= Nn) return;
    const float* zr = z + (size_t)wid * HID;
    float a0 = zr[lane], a1 = zr[lane + 32];
    float mx = fmaxf(a0, a1);
    #pragma unroll
    for (int o = 16; o > 0; o >>= 1) mx = fmaxf(mx, __shfl_xor(mx, o));
    float e0 = __expf(a0 - mx), e1 = __expf(a1 - mx);
    float s = e0 + e1;
    #pragma unroll
    for (int o = 16; o > 0; o >>= 1) s += __shfl_xor(s, o);
    float inv = 1.0f / s;
    R[(size_t)wid * HID + lane]      += e0 * inv;
    R[(size_t)wid * HID + lane + 32] += e1 * inv;
}

// ------------- global add pool over graphs ---------------------------------
__global__ void graph_pool(const float* __restrict__ R, const int* __restrict__ batch,
                           float* __restrict__ g, int Nn) {
    int t = blockIdx.x * blockDim.x + threadIdx.x;
    int total = Nn * HID;
    if (t >= total) return;
    int i = t >> 6, f = t & 63;
    atomicAdd(&g[(size_t)batch[i] * HID + f], R[t]);
}

// ------------- batchnorm (training-mode batch stats over GG rows) ----------
__global__ void bn_stats(const float* __restrict__ z, float* __restrict__ stats, int D) {
    int c = blockIdx.x * blockDim.x + threadIdx.x;
    if (c >= D) return;
    float s = 0.0f, s2 = 0.0f;
    for (int i = 0; i < GG; ++i) {
        float v = z[(size_t)i * D + c];
        s += v; s2 += v * v;
    }
    float m  = s * (1.0f / GG);
    float vr = s2 * (1.0f / GG) - m * m;     // biased variance
    stats[c]     = m;
    stats[D + c] = rsqrtf(vr + EPSI);
}

// normalize and emit bf16 (next consumer is a WMMA GEMM or bf16 head)
__global__ void bn_apply_bf16(const float* __restrict__ z, const float* __restrict__ stats,
                              const float* __restrict__ gamma, const float* __restrict__ beta,
                              __bf16* __restrict__ out, int D) {
    int t = blockIdx.x * blockDim.x + threadIdx.x;
    if (t >= GG * D) return;
    int c = t % D;
    out[t] = f2bf((z[t] - stats[c]) * stats[D + c] * gamma[c] + beta[c]);
}

// ------------- final 64 -> 2 layer + sigmoid -------------------------------
__global__ void head(const __bf16* __restrict__ h, const float* __restrict__ L4,
                     const float* __restrict__ lb4, float* __restrict__ out) {
    int t = blockIdx.x * blockDim.x + threadIdx.x;
    if (t >= GG * 2) return;
    int i = t >> 1, c = t & 1;
    float s = lb4[c];
    #pragma unroll 8
    for (int k = 0; k < HID; ++k) s += bf2f(h[(size_t)i * HID + k]) * L4[k * 2 + c];
    out[t] = 1.0f / (1.0f + __expf(-s));
}

// ---------------------------------------------------------------------------
static inline void launch_gemm(const __bf16* A, const __bf16* Bp, const float* bias,
                               float* C, int M, int Nd, int K, int flags,
                               hipStream_t stream) {
    dim3 grid((M + 63) / 64, Nd / 64);
    wmma_gemm<<<grid, 128, 0, stream>>>(A, Bp, bias, C, M, Nd, K, flags);
}

extern "C" void kernel_launch(void* const* d_in, const int* in_sizes, int n_in,
                              void* d_out, int out_size, void* d_ws, size_t ws_size,
                              hipStream_t stream) {
    const float* x     = (const float*)d_in[0];
    const int*   ei    = (const int*)d_in[1];
    const int*   batch = (const int*)d_in[2];
    const int E  = in_sizes[1] / 2;
    const int Nn = in_sizes[2];
    const int F  = in_sizes[0] / Nn;         // 128
    const int* row = ei;
    const int* col = ei + E;

    const float* W[4]  = {(const float*)d_in[4],  (const float*)d_in[6],
                          (const float*)d_in[8],  (const float*)d_in[10]};
    const float* bb[4] = {(const float*)d_in[5],  (const float*)d_in[7],
                          (const float*)d_in[9],  (const float*)d_in[11]};
    const float* Pw[4] = {(const float*)d_in[12], (const float*)d_in[14],
                          (const float*)d_in[16], (const float*)d_in[18]};
    const float* pb[4] = {(const float*)d_in[13], (const float*)d_in[15],
                          (const float*)d_in[17], (const float*)d_in[19]};
    const float* L1  = (const float*)d_in[20]; const float* lb1 = (const float*)d_in[21];
    const float* L2  = (const float*)d_in[22]; const float* lb2 = (const float*)d_in[23];
    const float* L3  = (const float*)d_in[24]; const float* lb3 = (const float*)d_in[25];
    const float* L4  = (const float*)d_in[26]; const float* lb4 = (const float*)d_in[27];
    const float* gam[3] = {(const float*)d_in[28], (const float*)d_in[30], (const float*)d_in[32]};
    const float* bet[3] = {(const float*)d_in[29], (const float*)d_in[31], (const float*)d_in[33]};

    // ---- workspace layout (256B-aligned carve-out) ----
    char* base = (char*)d_ws;
    size_t off = 0;
    auto alloc = [&](size_t bytes) -> void* {
        void* p = base + off;
        off += (bytes + 255) & ~(size_t)255;
        return p;
    };
    float*    dinv  = (float*)   alloc((size_t)Nn * 4);
    float*    R     = (float*)   alloc((size_t)Nn * HID * 4);
    float*    bufA  = (float*)   alloc((size_t)Nn * HID * 4);
    float*    bufB  = (float*)   alloc((size_t)Nn * HID * 4);
    unsigned* keys  = (unsigned*)alloc((size_t)Nn * HID * 4);
    __bf16*   xbf   = (__bf16*)  alloc((size_t)Nn * F * 2);
    __bf16*   hbf   = (__bf16*)  alloc((size_t)Nn * HID * 2);
    __bf16*   bpack = (__bf16*)  alloc((size_t)256 * 128 * 2);   // largest W: 256x128
    float*    gbuf  = (float*)   alloc((size_t)GG * HID * 4);
    __bf16*   gbf   = (__bf16*)  alloc((size_t)GG * HID * 2);
    float*    m1    = (float*)   alloc((size_t)GG * 256 * 4);
    __bf16*   m1b   = (__bf16*)  alloc((size_t)GG * 256 * 2);
    float*    m2    = (float*)   alloc((size_t)GG * 128 * 4);
    __bf16*   m2b   = (__bf16*)  alloc((size_t)GG * 128 * 2);
    float*    m3    = (float*)   alloc((size_t)GG * 64 * 4);
    __bf16*   m3b   = (__bf16*)  alloc((size_t)GG * 64 * 2);
    float*    stats = (float*)   alloc(512 * 4);

    const int B256 = 256;
    const int nodeFeat = Nn * HID;
    const int edgeFeat = (E + Nn) * HID;
    auto blk = [](int n, int b) { return (n + b - 1) / b; };

    // ---- degree / normalization: deg = 1 (self loop) + in-degree ----
    fill_f32<<<blk(Nn, B256), B256, 0, stream>>>(dinv, 1.0f, Nn);
    deg_add<<<blk(E, B256), B256, 0, stream>>>(col, dinv, E);
    deg_rsqrt<<<blk(Nn, B256), B256, 0, stream>>>(dinv, Nn);

    // ---- readout accumulator + input conversion ----
    fill_f32<<<blk(nodeFeat, B256), B256, 0, stream>>>(R, 0.0f, nodeFeat);
    cvt_f32_bf16<<<blk(Nn * F, B256), B256, 0, stream>>>(x, xbf, Nn * F);

    // ---- 4 GCN layers ----
    for (int l = 0; l < 4; ++l) {
        const __bf16* hin = (l == 0) ? xbf : hbf;
        const int K = (l == 0) ? F : HID;
        // h @ W  (WMMA)
        pack_Bfrag<<<blk(K * HID, B256), B256, 0, stream>>>(W[l], bpack, K, HID);
        launch_gemm(hin, bpack, nullptr, bufA, Nn, HID, K, 0, stream);
        // segment_sum(h[row] * norm, col)
        fill_f32<<<blk(nodeFeat, B256), B256, 0, stream>>>(bufB, 0.0f, nodeFeat);
        scatter_add<<<blk(edgeFeat, B256), B256, 0, stream>>>(bufA, row, col, dinv, bufB, E, Nn);
        // + bias, SELU
        selu_bias<<<blk(nodeFeat, B256), B256, 0, stream>>>(bufB, bb[l], nodeFeat);
        // segment_max(h[row], col) -> bf16 pooled features
        fill_f32<<<blk(nodeFeat, B256), B256, 0, stream>>>((float*)keys, 0.0f, nodeFeat);
        scatter_max<<<blk(edgeFeat, B256), B256, 0, stream>>>(bufB, row, col, keys, E, Nn);
        decode_max_bf16<<<blk(nodeFeat, B256), B256, 0, stream>>>(keys, hbf, nodeFeat);
        // readout: softmax(h @ P + pb) accumulated into R  (WMMA)
        pack_Bfrag<<<blk(HID * HID, B256), B256, 0, stream>>>(Pw[l], bpack, HID, HID);
        launch_gemm(hbf, bpack, pb[l], bufB, Nn, HID, HID, 1, stream);
        softmax_acc<<<blk(Nn * 32, B256), B256, 0, stream>>>(bufB, R, Nn);
    }

    // ---- global add pool over graphs ----
    fill_f32<<<blk(GG * HID, B256), B256, 0, stream>>>(gbuf, 0.0f, GG * HID);
    graph_pool<<<blk(nodeFeat, B256), B256, 0, stream>>>(R, batch, gbuf, Nn);
    cvt_f32_bf16<<<blk(GG * HID, B256), B256, 0, stream>>>(gbuf, gbf, GG * HID);

    // ---- MLP head: (gemm+relu, BN) x3, then 64->2 sigmoid ----
    pack_Bfrag<<<blk(HID * 256, B256), B256, 0, stream>>>(L1, bpack, HID, 256);
    launch_gemm(gbf, bpack, lb1, m1, GG, 256, HID, 3, stream);
    bn_stats<<<blk(256, B256), B256, 0, stream>>>(m1, stats, 256);
    bn_apply_bf16<<<blk(GG * 256, B256), B256, 0, stream>>>(m1, stats, gam[0], bet[0], m1b, 256);

    pack_Bfrag<<<blk(256 * 128, B256), B256, 0, stream>>>(L2, bpack, 256, 128);
    launch_gemm(m1b, bpack, lb2, m2, GG, 128, 256, 3, stream);
    bn_stats<<<blk(128, B256), B256, 0, stream>>>(m2, stats, 128);
    bn_apply_bf16<<<blk(GG * 128, B256), B256, 0, stream>>>(m2, stats, gam[1], bet[1], m2b, 128);

    pack_Bfrag<<<blk(128 * 64, B256), B256, 0, stream>>>(L3, bpack, 128, 64);
    launch_gemm(m2b, bpack, lb3, m3, GG, 64, 128, 3, stream);
    bn_stats<<<blk(64, B256), B256, 0, stream>>>(m3, stats, 64);
    bn_apply_bf16<<<blk(GG * 64, B256), B256, 0, stream>>>(m3, stats, gam[2], bet[2], m3b, 64);

    head<<<blk(GG * 2, B256), B256, 0, stream>>>(m3b, L4, lb4, (float*)d_out);
}
